// DownSampleToken_7524782703187
// MI455X (gfx1250) — matrix operand order repaired and checked by reading, not compiled, not run on hardware
//
#include <hip/hip_runtime.h>
#include <math.h>

// ---------------- problem constants ----------------
#define B_   4
#define C_   128
#define N_   4096
#define O_   128
#define NB_  4
#define TVALID_ (N_ + NB_)   // 4100 real key columns
#define TP_  4128            // padded to multiple of 32 (WMMA K granularity)
#define MT_  1024
#define KNN_ 32

// ---------------- WMMA plumbing (gfx1250, wave32) ----------------
typedef __attribute__((ext_vector_type(16))) __bf16 v16bf;
typedef __attribute__((ext_vector_type(8)))  float  v8f;

union ABf { v16bf v; unsigned short s[16]; };

__device__ __forceinline__ unsigned short f32_to_bf16(float f) {
  unsigned int u = __float_as_uint(f);
  u += 0x7FFFu + ((u >> 16) & 1u);          // round-to-nearest-even
  return (unsigned short)(u >> 16);
}

// A operand (16x32 bf16, ISA 7.12.2): lane m=lane&15, g=lane>>4.
// elements 0..7  = K[g*8   .. g*8+7], elements 8..15 = K[16+g*8 .. 16+g*8+7]
__device__ __forceinline__ v16bf load_bf16_A(const unsigned short* rowPtr, int k0, int g) {
  ABf a;
  *(uint4*)(&a.s[0]) = *(const uint4*)(rowPtr + k0 + g * 8);
  *(uint4*)(&a.s[8]) = *(const uint4*)(rowPtr + k0 + 16 + g * 8);
  return a.v;
}
// B operand (32x16 bf16): lane n=lane&15 is the output column; g=lane>>4,
// elements 0..7 = K[g*8 .. g*8+7], elements 8..15 = K[16+g*8 .. 16+g*8+7].
// rowPtr points at the "column row" (buffer storing B columns as contiguous rows).
__device__ __forceinline__ v16bf load_bf16_B(const unsigned short* rowPtr, int k0, int g) {
  ABf b;
  *(uint4*)(&b.s[0]) = *(const uint4*)(rowPtr + k0 + g * 8);
  *(uint4*)(&b.s[8]) = *(const uint4*)(rowPtr + k0 + 16 + g * 8);
  return b.v;
}

#define WMMA_BF16(acc, Av, Bv) \
  (acc) = __builtin_amdgcn_wmma_f32_16x16x32_bf16(false, (Av), false, (Bv), (short)0, (acc), false, false)

// gfx1250 async copy global->LDS (ASYNCcnt path); 16B per lane.
__device__ __forceinline__ void async_ld_b128(unsigned int lds_byte_addr, const void* gsrc) {
  asm volatile("global_load_async_to_lds_b128 %0, %1, off"
               :: "v"(lds_byte_addr), "v"((unsigned long long)(size_t)gsrc)
               : "memory");
}
__device__ __forceinline__ void wait_async_le2() {
  asm volatile("s_wait_asynccnt 0x2" ::: "memory");
}
__device__ __forceinline__ void wait_async_0() {
  asm volatile("s_wait_asynccnt 0x0" ::: "memory");
}
__device__ __forceinline__ unsigned int lds_addr_of(const void* p) {
  // generic pointers into LDS carry the LDS byte offset in addr[31:0]
  return (unsigned int)(unsigned long long)(size_t)p;
}

// D layout: element e of lane L is D[(L>>4)*8 + e][L&15]

// ---------------- small utility kernels ----------------
__global__ void zero_f32_kernel(float* p, int n) {
  int i = blockIdx.x * blockDim.x + threadIdx.x;
  if (i < n) p[i] = 0.0f;
}

__global__ void cvt_bf16_kernel(const float* __restrict__ s, unsigned short* __restrict__ d, int n) {
  int i = blockIdx.x * blockDim.x + threadIdx.x;
  if (i < n) d[i] = f32_to_bf16(s[i]);
}

// xtT[b][t][c] = xt[b][c][t] in bf16; rows N..N+3 are bin tokens, rest zero pad.
__global__ void build_xtT_kernel(const float* __restrict__ x, const float* __restrict__ tok,
                                 unsigned short* __restrict__ xtT) {
  __shared__ float tile[C_][33];
  int b  = blockIdx.y;
  int t0 = blockIdx.x * 32;
  int tx = threadIdx.x & 31, ty = threadIdx.x >> 5;
  int t  = t0 + tx;
  for (int c = ty; c < C_; c += 8) {
    float v = 0.0f;
    if (t < N_)           v = x[((size_t)b * C_ + c) * N_ + t];
    else if (t < TVALID_) v = tok[c * NB_ + (t - N_)];
    tile[c][tx] = v;
  }
  __syncthreads();
  for (int idx = threadIdx.x; idx < 32 * C_; idx += 256) {
    int tt = idx >> 7, c = idx & (C_ - 1);
    xtT[((size_t)b * TP_ + (t0 + tt)) * C_ + c] = f32_to_bf16(tile[c][tt]);
  }
}

__global__ void x2_kernel(const float* __restrict__ x, float* __restrict__ x2) {
  int i = blockIdx.x * blockDim.x + threadIdx.x;
  if (i >= B_ * N_) return;
  int b = i / N_, n = i - b * N_;
  const float* p = x + (size_t)b * C_ * N_ + n;
  float s = 0.0f;
  for (int c = 0; c < C_; ++c) { float v = p[(size_t)c * N_]; s += v * v; }
  x2[i] = s;
}

// ---------------- fused QKV projection (3 GEMMs share the A operand) ----------
// q16[b][n][o] (bf16), k16[b][t][o] (bf16), v_om[b][o][t] (bf16, = v in 'bot')
__global__ void qkv_kernel(const unsigned short* __restrict__ xtT,
                           const unsigned short* __restrict__ Wq16,
                           const unsigned short* __restrict__ Wk16,
                           const unsigned short* __restrict__ Wv16,
                           unsigned short* __restrict__ q16,
                           unsigned short* __restrict__ k16,
                           unsigned short* __restrict__ v_om) {
  const int nt = TP_ / 16;
  int b    = blockIdx.x / nt;
  int t0   = (blockIdx.x % nt) * 16;
  int wave = threadIdx.x >> 5, lane = threadIdx.x & 31;
  int nl = lane & 15, g = lane >> 4;
  int o0 = wave * 16;                       // 8 waves -> 8 o-tiles = O_=128
  const unsigned short* arow = xtT + ((size_t)b * TP_ + t0 + nl) * C_;
  const unsigned short* bq = Wq16 + (size_t)(o0 + nl) * C_;
  const unsigned short* bk = Wk16 + (size_t)(o0 + nl) * C_;
  const unsigned short* bv = Wv16 + (size_t)(o0 + nl) * C_;
  v8f aq = {}, ak = {}, av = {};
#pragma unroll
  for (int k0 = 0; k0 < C_; k0 += 32) {
    v16bf A = load_bf16_A(arow, k0, g);
    WMMA_BF16(aq, A, load_bf16_B(bq, k0, g));
    WMMA_BF16(ak, A, load_bf16_B(bk, k0, g));
    WMMA_BF16(av, A, load_bf16_B(bv, k0, g));
  }
  int oc = o0 + nl;
#pragma unroll
  for (int e = 0; e < 8; ++e) {
    int t = t0 + g * 8 + e;
    if (t < N_) q16[((size_t)b * N_ + t) * O_ + oc] = f32_to_bf16(aq[e]);
    k16[((size_t)b * TP_ + t) * O_ + oc]  = f32_to_bf16(ak[e]);
    v_om[((size_t)b * O_ + oc) * TP_ + t] = f32_to_bf16(av[e]);
  }
}

// ---------------- Gram matrix + per-row top-32 (16-row slab in 256KB LDS) ----
__global__ void gram_topk_kernel(const unsigned short* __restrict__ xtT,
                                 const float* __restrict__ x2,
                                 int* __restrict__ nn_idx,
                                 float* __restrict__ sparse_num) {
  extern __shared__ float dist[];            // [16][N_]  = 256 KB
  const int nblk = N_ / 16;
  int b  = blockIdx.x / nblk;
  int i0 = (blockIdx.x % nblk) * 16;
  int wave = threadIdx.x >> 5, lane = threadIdx.x & 31;   // 16 waves
  int nl = lane & 15, g = lane >> 4;
  const unsigned short* xb = xtT + (size_t)b * TP_ * C_;
  const unsigned short* arow = xb + (size_t)(i0 + nl) * C_;
  const float* x2b = x2 + (size_t)b * N_;
  for (int jt = wave; jt < N_ / 16; jt += 16) {
    int j0 = jt * 16;
    const unsigned short* brow = xb + (size_t)(j0 + nl) * C_;
    __builtin_prefetch(brow + 16 * 16 * C_, 0, 0);
    v8f acc = {};
#pragma unroll
    for (int k0 = 0; k0 < C_; k0 += 32)
      WMMA_BF16(acc, load_bf16_A(arow, k0, g), load_bf16_B(brow, k0, g));
    int j = j0 + nl;
    float x2j = x2b[j];
#pragma unroll
    for (int e = 0; e < 8; ++e) {
      int row = g * 8 + e;
      dist[(size_t)row * N_ + j] = x2b[i0 + row] + x2j - 2.0f * acc[e];
    }
  }
  __syncthreads();
  // wave w selects top-32 (smallest dist, ties -> lowest index) for row w
  int row = wave;
  float* drow = dist + (size_t)row * N_;
  size_t gRow = (size_t)b * N_ + i0 + row;
  for (int it = 0; it < KNN_; ++it) {
    float best = 3.0e38f; int bidx = 0x7fffffff;
    for (int j = lane; j < N_; j += 32) {
      float v = drow[j];
      if (v < best) { best = v; bidx = j; }
    }
#pragma unroll
    for (int off = 16; off > 0; off >>= 1) {
      float ov = __shfl_xor(best, off, 32);
      int   oi = __shfl_xor(bidx, off, 32);
      if (ov < best || (ov == best && oi < bidx)) { best = ov; bidx = oi; }
    }
    // butterfly made best/bidx uniform; the owning lane (its stripe) kills it
    if (lane == (bidx & 31)) drow[bidx] = 3.0e38f;
    if (lane == 0) {
      nn_idx[gRow * KNN_ + it] = bidx;
      atomicAdd(&sparse_num[(size_t)b * N_ + bidx], 1.0f);
    }
  }
}

// ---------------- attention pass 1: softmax stats + NN/bin extraction -------
__global__ void attn_stats_kernel(const unsigned short* __restrict__ q16,
                                  const unsigned short* __restrict__ k16,
                                  const int* __restrict__ nn_idx,
                                  float* __restrict__ aps_num,
                                  float* __restrict__ row_max,
                                  float* __restrict__ row_sum,
                                  unsigned int* __restrict__ bin_bits) {
  extern __shared__ float lg[];              // [16][TP_] = 264 KB logits slab
  __shared__ float red[16][18];
  const int nblk = N_ / 16;
  int b  = blockIdx.x / nblk;
  int n0 = (blockIdx.x % nblk) * 16;
  int wave = threadIdx.x >> 5, lane = threadIdx.x & 31;   // 8 waves
  int nl = lane & 15, g = lane >> 4;
  const unsigned short* qrow = q16 + ((size_t)b * N_ + n0 + nl) * O_;
  const unsigned short* kb   = k16 + (size_t)b * TP_ * O_;
  for (int ct = wave; ct < TP_ / 16; ct += 8) {
    int t0 = ct * 16;
    const unsigned short* brow = kb + (size_t)(t0 + nl) * O_;
    __builtin_prefetch(brow + 8 * 16 * O_, 0, 0);
    v8f acc = {};
#pragma unroll
    for (int k0 = 0; k0 < O_; k0 += 32)
      WMMA_BF16(acc, load_bf16_A(qrow, k0, g), load_bf16_B(brow, k0, g));
    int t = t0 + nl;
    bool valid = (t < TVALID_);
#pragma unroll
    for (int e = 0; e < 8; ++e) {
      int row = g * 8 + e;
      lg[(size_t)row * TP_ + t] = valid ? acc[e] : -3.0e38f;
    }
  }
  __syncthreads();
  // 16 groups of 16 threads: softmax stats per row (f32 exact over 4100 cols)
  int grp = threadIdx.x >> 4;
  int sub = threadIdx.x & 15;
  float* r = lg + (size_t)grp * TP_;
  float mx = -3.0e38f;
  for (int t = sub; t < TVALID_; t += 16) mx = fmaxf(mx, r[t]);
  red[grp][sub] = mx;
  __syncthreads();
  if (sub == 0) {
    float v = red[grp][0];
    for (int i = 1; i < 16; ++i) v = fmaxf(v, red[grp][i]);
    red[grp][16] = v;
  }
  __syncthreads();
  float mrow = red[grp][16];
  float sm = 0.0f;
  for (int t = sub; t < TVALID_; t += 16) sm += __expf(r[t] - mrow);
  red[grp][sub] = sm;
  __syncthreads();
  if (sub == 0) {
    float v = 0.0f;
    for (int i = 0; i < 16; ++i) v += red[grp][i];
    red[grp][17] = v;
  }
  __syncthreads();
  float srow = red[grp][17];
  size_t gn = (size_t)b * N_ + n0 + grp;
  if (sub == 0) { row_max[gn] = mrow; row_sum[gn] = srow; }
  if (sub < NB_) {   // bin_prob = max over rows of attn[:, N+bin]
    float bv = __expf(r[N_ + sub] - mrow) / srow;
    atomicMax(&bin_bits[b * NB_ + sub], __float_as_uint(bv)); // non-neg floats
  }
  const int* nn = nn_idx + gn * KNN_;        // aps numerator scatter
  for (int it = sub; it < KNN_; it += 16) {
    int j = nn[it];
    float a = __expf(r[j] - mrow) / srow;
    atomicAdd(&aps_num[(size_t)b * N_ + j], a);
  }
}

__global__ void aps_kernel(const float* __restrict__ aps_num,
                           const float* __restrict__ sparse_num,
                           float* __restrict__ aps) {
  int i = blockIdx.x * blockDim.x + threadIdx.x;
  if (i < B_ * N_) {
    float sn = sparse_num[i] + 1e-8f;
    aps[i] = aps_num[i] / (sn * sn);
  }
}

// ---------------- descending argsort per batch (bitonic in LDS) -------------
__global__ void argsort_kernel(const float* __restrict__ aps, int* __restrict__ order) {
  __shared__ float key[N_];
  __shared__ int   idx[N_];
  int b = blockIdx.x;
  for (int i = threadIdx.x; i < N_; i += blockDim.x) { key[i] = aps[(size_t)b * N_ + i]; idx[i] = i; }
  __syncthreads();
  for (int k = 2; k <= N_; k <<= 1) {
    for (int j = k >> 1; j > 0; j >>= 1) {
      for (int i = threadIdx.x; i < N_; i += blockDim.x) {
        int l = i ^ j;
        if (l > i) {
          bool up = ((i & k) == 0);
          float ka = key[i], kb = key[l];
          int   ia = idx[i], ib = idx[l];
          bool aFirst = (ka > kb) || (ka == kb && ia < ib); // stable desc
          if (up ? !aFirst : aFirst) { key[i] = kb; key[l] = ka; idx[i] = ib; idx[l] = ia; }
        }
      }
      __syncthreads();
    }
  }
  for (int i = threadIdx.x; i < N_; i += blockDim.x) order[(size_t)b * N_ + i] = idx[i];
}

// ---------------- bin bookkeeping -> idx_down ------------------------------
__global__ void select_kernel(const int* __restrict__ order,
                              const unsigned int* __restrict__ bin_bits,
                              int* __restrict__ idx_down,
                              float* __restrict__ out_idx) {
  __shared__ int cum[NB_], cumprev[NB_];
  int b = blockIdx.x;
  if (threadIdx.x == 0) {
    int kb[NB_];
    for (int i = 0; i < NB_; ++i) {
      float bp = __uint_as_float(bin_bits[b * NB_ + i]);
      kb[i] = (int)floorf((2.0f * MT_ / NB_) * bp);
    }
    kb[NB_ - 1] = MT_ - (kb[0] + kb[1] + kb[2]);
    int c = 0;
    for (int i = 0; i < NB_; ++i) { cumprev[i] = c; c += kb[i]; cum[i] = c; }
  }
  __syncthreads();
  for (int m = threadIdx.x; m < MT_; m += blockDim.x) {
    int bo = 0;
    for (int i = 0; i < NB_; ++i) bo += (m >= cum[i]) ? 1 : 0;
    if (bo > NB_ - 1) bo = NB_ - 1;
    int off = m - cumprev[bo];
    int sp  = bo * (N_ / NB_) + off;
    int id  = order[(size_t)b * N_ + sp];
    idx_down[(size_t)b * MT_ + m] = id;
    out_idx[(size_t)b * MT_ + m]  = (float)id;
  }
}

// ---------------- recompute selected rows, P staged bf16 in LDS, P @ v^T ----
// Phase 2 streams the shared v^T K-slab (128 o-rows x 32 t = 8 KB/step) into
// LDS with async-to-LDS DMA, double buffered on ASYNCcnt, so all 8 waves pull
// B operands from LDS while the next slab is in flight.
#define KSTAGE_ (128 * 32)                   // ushorts per buffer
__global__ void downsample_kernel(const unsigned short* __restrict__ q16,
                                  const unsigned short* __restrict__ k16,
                                  const unsigned short* __restrict__ v_om,
                                  const int* __restrict__ idx_down,
                                  const float* __restrict__ row_max,
                                  const float* __restrict__ row_sum,
                                  float* __restrict__ out) {   // (B,O,MT)
  extern __shared__ unsigned short P[];      // [16][TP_] bf16 = 132 KB, then 2x8KB stage
  unsigned short* kstage = P + (size_t)16 * TP_;
  __shared__ int   rowsel[16];
  __shared__ float rmax[16], rsum[16];
  const int nblk = MT_ / 16;
  int b  = blockIdx.x / nblk;
  int m0 = (blockIdx.x % nblk) * 16;
  if (threadIdx.x < 16) {
    int id = idx_down[(size_t)b * MT_ + m0 + threadIdx.x];
    rowsel[threadIdx.x] = id;
    rmax[threadIdx.x] = row_max[(size_t)b * N_ + id];
    rsum[threadIdx.x] = row_sum[(size_t)b * N_ + id];
  }
  __syncthreads();
  int wave = threadIdx.x >> 5, lane = threadIdx.x & 31;   // 8 waves
  int nl = lane & 15, g = lane >> 4;
  // phase 1: logits for the 16 selected rows -> normalized P (bf16) in LDS
  const unsigned short* qrow = q16 + ((size_t)b * N_ + rowsel[nl]) * O_;
  const unsigned short* kb   = k16 + (size_t)b * TP_ * O_;
  for (int ct = wave; ct < TP_ / 16; ct += 8) {
    int t0 = ct * 16;
    const unsigned short* brow = kb + (size_t)(t0 + nl) * O_;
    __builtin_prefetch(brow + 8 * 16 * O_, 0, 0);
    v8f acc = {};
#pragma unroll
    for (int k0 = 0; k0 < O_; k0 += 32)
      WMMA_BF16(acc, load_bf16_A(qrow, k0, g), load_bf16_B(brow, k0, g));
    int t = t0 + nl;
    bool valid = (t < TVALID_);
#pragma unroll
    for (int e = 0; e < 8; ++e) {
      int row = g * 8 + e;
      float pv = valid ? __expf(acc[e] - rmax[row]) / rsum[row] : 0.0f;
      P[(size_t)row * TP_ + t] = f32_to_bf16(pv);
    }
  }
  __syncthreads();
  // phase 2: D[m][o-tile(wave)] = P(16 x TP_) @ v^T, K swept 32 at a time.
  // Stage v_om[0..127][t0..t0+31] (8 KB) per step via async DMA to LDS.
  const unsigned short* vb = v_om + (size_t)b * O_ * TP_;
  int o0 = wave * 16;
  const unsigned short* arow = P + (size_t)nl * TP_;
  const int nsteps = TP_ / 32;               // 129
  // prime buffer 0
  {
    unsigned short* dst = kstage;
    for (int ci = threadIdx.x; ci < 512; ci += 256) {  // 2 async ops per wave
      int o = ci >> 2, part = ci & 3;
      async_ld_b128(lds_addr_of(dst + o * 32 + part * 8),
                    vb + (size_t)o * TP_ + part * 8);
    }
  }
  v8f acc = {};
  for (int s = 0; s < nsteps; ++s) {
    int cur = s & 1;
    if (s + 1 < nsteps) {                    // stage next slab into other buffer
      unsigned short* dst = kstage + (cur ^ 1) * KSTAGE_;
      int t0n = (s + 1) * 32;
      for (int ci = threadIdx.x; ci < 512; ci += 256) {
        int o = ci >> 2, part = ci & 3;
        async_ld_b128(lds_addr_of(dst + o * 32 + part * 8),
                      vb + (size_t)o * TP_ + t0n + part * 8);
      }
      wait_async_le2();                      // current slab's 2 ops retired
    } else {
      wait_async_0();
    }
    __syncthreads();                         // slab visible to all waves
    const unsigned short* brow = kstage + cur * KSTAGE_ + (size_t)(o0 + nl) * 32;
    WMMA_BF16(acc, load_bf16_A(arow, s * 32, g), load_bf16_B(brow, 0, g));
    __syncthreads();                         // done reading before next overwrite
  }
  int oc = o0 + nl;
#pragma unroll
  for (int e = 0; e < 8; ++e) {
    int m = g * 8 + e;
    out[((size_t)b * O_ + oc) * MT_ + m0 + m] = acc[e];   // contiguous 32B/lane
  }
}

// ---------------- host-side orchestration ----------------------------------
extern "C" void kernel_launch(void* const* d_in, const int* in_sizes, int n_in,
                              void* d_out, int out_size, void* d_ws, size_t ws_size,
                              hipStream_t stream) {
  (void)in_sizes; (void)n_in; (void)out_size; (void)ws_size;
  const float* x   = (const float*)d_in[0];
  const float* tok = (const float*)d_in[1];
  const float* Wq  = (const float*)d_in[2];
  const float* Wk  = (const float*)d_in[3];
  const float* Wv  = (const float*)d_in[4];
  float* out_xds = (float*)d_out;                       // (B,O,MT) fp32
  float* out_idx = out_xds + (size_t)B_ * O_ * MT_;     // (B,1,MT)

  // workspace carving (~20 MB total)
  char* w = (char*)d_ws;
  auto alloc = [&](size_t bytes) -> void* {
    void* p = (void*)w; w += (bytes + 255) & ~(size_t)255; return p;
  };
  unsigned short* xtT  = (unsigned short*)alloc((size_t)B_ * TP_ * C_ * 2);
  unsigned short* Wq16 = (unsigned short*)alloc((size_t)O_ * C_ * 2);
  unsigned short* Wk16 = (unsigned short*)alloc((size_t)O_ * C_ * 2);
  unsigned short* Wv16 = (unsigned short*)alloc((size_t)O_ * C_ * 2);
  unsigned short* q16  = (unsigned short*)alloc((size_t)B_ * N_ * O_ * 2);
  unsigned short* k16  = (unsigned short*)alloc((size_t)B_ * TP_ * O_ * 2);
  unsigned short* v_om = (unsigned short*)alloc((size_t)B_ * O_ * TP_ * 2);
  float* x2         = (float*)alloc((size_t)B_ * N_ * 4);
  int*   nn_idx     = (int*)alloc((size_t)B_ * N_ * KNN_ * 4);
  // the next three are zeroed with one launch (allocations stay contiguous)
  float* sparse_num = (float*)alloc((size_t)B_ * N_ * 4);
  float* aps_num    = (float*)alloc((size_t)B_ * N_ * 4);
  unsigned int* bin_bits = (unsigned int*)alloc(256);
  float* row_max    = (float*)alloc((size_t)B_ * N_ * 4);
  float* row_sum    = (float*)alloc((size_t)B_ * N_ * 4);
  float* aps        = (float*)alloc((size_t)B_ * N_ * 4);
  int*   order      = (int*)alloc((size_t)B_ * N_ * 4);
  int*   idx_down   = (int*)alloc((size_t)B_ * MT_ * 4);

  // big-LDS kernels: 16-row fp32/bf16 slabs exploit CDNA5's 320KB-per-WGP LDS
  const int ds_lds = 16 * TP_ * 2 + 2 * KSTAGE_ * 2;   // P slab + 2 async buffers
  (void)hipFuncSetAttribute((const void*)gram_topk_kernel,
      hipFuncAttributeMaxDynamicSharedMemorySize, 16 * N_ * 4);
  (void)hipFuncSetAttribute((const void*)attn_stats_kernel,
      hipFuncAttributeMaxDynamicSharedMemorySize, 16 * TP_ * 4);
  (void)hipFuncSetAttribute((const void*)downsample_kernel,
      hipFuncAttributeMaxDynamicSharedMemorySize, ds_lds);

  int zc = 2 * B_ * N_ + 64;
  zero_f32_kernel<<<(zc + 255) / 256, 256, 0, stream>>>(sparse_num, zc);
  cvt_bf16_kernel<<<(O_ * C_ + 255) / 256, 256, 0, stream>>>(Wq, Wq16, O_ * C_);
  cvt_bf16_kernel<<<(O_ * C_ + 255) / 256, 256, 0, stream>>>(Wk, Wk16, O_ * C_);
  cvt_bf16_kernel<<<(O_ * C_ + 255) / 256, 256, 0, stream>>>(Wv, Wv16, O_ * C_);
  build_xtT_kernel<<<dim3(TP_ / 32, B_), 256, 0, stream>>>(x, tok, xtT);
  x2_kernel<<<(B_ * N_ + 255) / 256, 256, 0, stream>>>(x, x2);
  qkv_kernel<<<B_ * (TP_ / 16), 256, 0, stream>>>(xtT, Wq16, Wk16, Wv16, q16, k16, v_om);
  gram_topk_kernel<<<B_ * (N_ / 16), 512, 16 * N_ * 4, stream>>>(xtT, x2, nn_idx, sparse_num);
  attn_stats_kernel<<<B_ * (N_ / 16), 256, 16 * TP_ * 4, stream>>>(
      q16, k16, nn_idx, aps_num, row_max, row_sum, bin_bits);
  aps_kernel<<<(B_ * N_ + 255) / 256, 256, 0, stream>>>(aps_num, sparse_num, aps);
  argsort_kernel<<<B_, 1024, 0, stream>>>(aps, order);
  select_kernel<<<B_, 256, 0, stream>>>(order, bin_bits, idx_down, out_idx);
  downsample_kernel<<<B_ * (MT_ / 16), 256, ds_lds, stream>>>(
      q16, k16, v_om, idx_down, row_max, row_sum, out_xds);
}
